// Part_Graph_13632226198264
// MI455X (gfx1250) — matrix operand order repaired and checked by reading, not compiled, not run on hardware
//
#include <hip/hip_runtime.h>
#include <hip/hip_bf16.h>

typedef __attribute__((ext_vector_type(16))) __bf16    v16bf;
typedef __attribute__((ext_vector_type(8)))  float     v8f;
typedef __attribute__((ext_vector_type(4)))  unsigned int v4u;
typedef int v4i_ __attribute__((vector_size(16)));

#define NB   16
#define HW   3600
#define CIN  256
#define HID  10
#define PTOT (NB*HW)          // 57600 positions
#define NT   48               // N (position) tile for fused GEMM
#define BNINV 0.9999950000374997f   // 1/sqrt(1+1e-5)

__device__ __forceinline__ unsigned short f2bf(float f) {
  unsigned int u = __builtin_bit_cast(unsigned int, f);
  unsigned int r = (u + 0x7FFFu + ((u >> 16) & 1u)) >> 16;
  return (unsigned short)r;
}

// ---- CDNA5 async global->LDS copy (ASYNCcnt path), with safe fallback ----
#if __has_builtin(__builtin_amdgcn_global_load_async_to_lds_b128)
#define HAVE_ASYNC_LDS 1
__device__ __forceinline__ void async_copy16(const unsigned short* g, unsigned short* l) {
  __builtin_amdgcn_global_load_async_to_lds_b128(
      (__attribute__((address_space(1))) v4i_*)g,
      (__attribute__((address_space(3))) v4i_*)l, 0, 0);
}
#else
#define HAVE_ASYNC_LDS 0
#endif

#if __has_builtin(__builtin_amdgcn_s_wait_asynccnt)
#define WAIT_ASYNC() __builtin_amdgcn_s_wait_asynccnt(0)
#else
#define WAIT_ASYNC() asm volatile("s_wait_asynccnt 0x0" ::: "memory")
#endif

// ---------------------------------------------------------------- convert
__global__ __launch_bounds__(256) void k_f32_to_bf16(
    const float* __restrict__ src, unsigned short* __restrict__ dst, int n) {
  int i = blockIdx.x * 256 + threadIdx.x;
  if (i < n) dst[i] = f2bf(src[i]);
}

// ------------------------------------------------- fused proj1+proj2 GEMM
// grid: 6 (head) * 16 (batch) * 75 (hw tiles of 48); block: 256 thr (8 waves)
__global__ __launch_bounds__(256) void k_proj_fused(
    const unsigned short* __restrict__ xpbf,   // [16][256][3600] bf16
    const unsigned short* __restrict__ w1bf,   // [6][256][256]  bf16
    const unsigned short* __restrict__ w2bf,   // [6][20][256]   bf16
    const float* __restrict__ g1, const float* __restrict__ bb1,   // [6][256]
    const float* __restrict__ g2, const float* __restrict__ bb2,   // [6][20]
    float* __restrict__ ctx)                   // [6][16][20][3600] f32
{
  int blk = blockIdx.x;
  int nt  = blk % 75;
  int n   = (blk / 75) % NB;
  int i   = blk / (75 * NB);
  int hw0 = nt * NT;

  __shared__ unsigned short sA1[256 * 32];   // proj1 weight tile  (16 KB)
  __shared__ unsigned short sB1[NT * 32];    // xp tile, transposed (3 KB)
  __shared__ unsigned short sH [NT * 256];   // h1 tile, K-major   (24 KB)
  __shared__ unsigned short sA2[32 * 256];   // proj2 weights pad32 (16 KB)

  int t    = threadIdx.x;
  int lane = t & 31;
  int wv   = t >> 5;

  // ---- load proj2 weights once (pad rows 20..31 with zero)
  {
    int r  = t >> 3;            // 0..31
    int c0 = (t & 7) * 32;      // 32 bf16 = 64B per thread
    v4u* d = (v4u*)&sA2[r * 256 + c0];
    if (r < 20) {
      const v4u* s = (const v4u*)&w2bf[(i * 20 + r) * 256 + c0];
      d[0] = s[0]; d[1] = s[1]; d[2] = s[2]; d[3] = s[3];
    } else {
      v4u z = {0u, 0u, 0u, 0u};
      d[0] = z; d[1] = z; d[2] = z; d[3] = z;
    }
  }

  const unsigned short* w1 = w1bf + i * 256 * 256;
  const unsigned short* xb = xpbf + n * 256 * HW;

  v8f acc[2][3] = {};

  // ---- stage 1: h1 = relu(bn(W1 @ xp)), all 256 out channels for 48 cols
  for (int kk = 0; kk < 8; ++kk) {
    int k0 = kk * 32;
    __syncthreads();
#if HAVE_ASYNC_LDS
    {   // A tile: thread t -> row t, 32 bf16 (64B) straight to LDS, async
      const unsigned short* g = &w1[t * 256 + k0];
      unsigned short* l = &sA1[t * 32];
      async_copy16(g,      l);
      async_copy16(g + 8,  l + 8);
      async_copy16(g + 16, l + 16);
      async_copy16(g + 24, l + 24);
    }
#else
    {   // A tile: thread t -> row t, 32 bf16 (64 bytes)
      const v4u* s = (const v4u*)&w1[t * 256 + k0];
      v4u* d = (v4u*)&sA1[t * 32];
      d[0] = s[0]; d[1] = s[1]; d[2] = s[2]; d[3] = s[3];
    }
#endif
    // B tile transposed into [col][k]
    for (int j = 0; j < 6; ++j) {
      int idx = t + 256 * j;                 // 0..1535
      int c = idx % NT, k = idx / NT;
      sB1[c * 32 + k] = xb[(k0 + k) * HW + hw0 + c];
    }
    // prefetch next K-step's B rows while WMMAs run
    if (kk < 7 && t < 32)
      __builtin_prefetch(&xb[(k0 + 32 + t) * HW + hw0], 0, 0);
#if HAVE_ASYNC_LDS
    WAIT_ASYNC();
#endif
    __syncthreads();

    int kh = (lane >> 4) * 8;      // A frag K-half select
    int kg = (lane >> 4) * 8;      // B frag K-half select
    union { v16bf v; v4u q[2]; } af[2], bfg[3];
    for (int mi = 0; mi < 2; ++mi) {
      int r = (wv + mi * 8) * 16 + (lane & 15);
      af[mi].q[0] = *(const v4u*)&sA1[r * 32 + kh];
      af[mi].q[1] = *(const v4u*)&sA1[r * 32 + 16 + kh];
    }
    for (int nj = 0; nj < 3; ++nj) {
      int c = nj * 16 + (lane & 15);
      bfg[nj].q[0] = *(const v4u*)&sB1[c * 32 + kg];
      bfg[nj].q[1] = *(const v4u*)&sB1[c * 32 + kg + 16];
    }
    for (int mi = 0; mi < 2; ++mi)
      for (int nj = 0; nj < 3; ++nj)
        acc[mi][nj] = __builtin_amdgcn_wmma_f32_16x16x32_bf16(
            false, af[mi].v, false, bfg[nj].v, (short)0, acc[mi][nj],
            false, false);
  }

  // ---- stage-1 epilogue: BN + ReLU -> packed bf16x2 into sH[col][ch]
  __syncthreads();
  {
    const float* gg = g1 + i * 256;
    const float* gb = bb1 + i * 256;
    unsigned int* sH32 = (unsigned int*)sH;
    for (int mi = 0; mi < 2; ++mi) {
      int chBase = (wv + mi * 8) * 16 + (lane >> 4) * 8;
      for (int nj = 0; nj < 3; ++nj) {
        int c = nj * 16 + (lane & 15);
        for (int v2 = 0; v2 < 4; ++v2) {
          int ch = chBase + 2 * v2;
          float x0 = acc[mi][nj][2 * v2]     * (gg[ch]     * BNINV) + gb[ch];
          float x1 = acc[mi][nj][2 * v2 + 1] * (gg[ch + 1] * BNINV) + gb[ch + 1];
          x0 = x0 > 0.f ? x0 : 0.f;
          x1 = x1 > 0.f ? x1 : 0.f;
          unsigned int pk = (unsigned int)f2bf(x0) | ((unsigned int)f2bf(x1) << 16);
          sH32[(c * 256 + ch) >> 1] = pk;
        }
      }
    }
  }
  __syncthreads();

  // ---- stage 2: ctx = relu(bn(W2 @ h1)), 20 (pad 32) out channels
  if (wv < 6) {
    int msub = wv & 1, nj2 = wv >> 1;
    v8f acc2 = {};
    for (int kk = 0; kk < 8; ++kk) {
      int k0 = kk * 32;
      int kh = (lane >> 4) * 8;
      int kg = (lane >> 4) * 8;
      union { v16bf v; v4u q[2]; } a2, b2;
      int r = msub * 16 + (lane & 15);
      a2.q[0] = *(const v4u*)&sA2[r * 256 + k0 + kh];
      a2.q[1] = *(const v4u*)&sA2[r * 256 + k0 + 16 + kh];
      int c = nj2 * 16 + (lane & 15);
      b2.q[0] = *(const v4u*)&sH[c * 256 + k0 + kg];
      b2.q[1] = *(const v4u*)&sH[c * 256 + k0 + kg + 16];
      acc2 = __builtin_amdgcn_wmma_f32_16x16x32_bf16(
          false, a2.v, false, b2.v, (short)0, acc2, false, false);
    }
    for (int v = 0; v < 8; ++v) {
      int oc = msub * 16 + (lane >> 4) * 8 + v;
      if (oc < 20) {
        float val = acc2[v] * (g2[i * 20 + oc] * BNINV) + bb2[i * 20 + oc];
        val = val > 0.f ? val : 0.f;
        int hw = hw0 + nj2 * 16 + (lane & 15);
        ctx[((i * NB + n) * 20 + oc) * HW + hw] = val;
      }
    }
  }
}

// ------------------------------------------------------------- heads
__global__ __launch_bounds__(256) void k_heads(
    const float* __restrict__ ctx,
    const float* __restrict__ xh0, const float* __restrict__ xh1,
    const float* __restrict__ att_w, const float* __restrict__ att_b,
    const float* __restrict__ ctx_w, const float* __restrict__ ctx_b,
    const float* __restrict__ du_w,  const float* __restrict__ du_b,
    const float* __restrict__ dl_w,  const float* __restrict__ dl_b,
    float* __restrict__ att_soft,    // ws [6][16][3][3600]
    float* __restrict__ pu_att,      // ws [16][5][3600]
    float* __restrict__ pl_att,      // ws [16][3][3600]
    float* __restrict__ out)
{
  int p = blockIdx.x * 256 + threadIdx.x;
  if (p >= PTOT) return;
  int n = p / HW, hw = p % HW;

  float* out_pu  = out + 3456000;
  float* out_pl  = out + 3744000;
  float* out_att = out + 3916800;
  float* out_ctx = out + 4838400;

  const int CNT[6]  = {2, 3, 3, 3, 3, 2};
  const int COFF[6] = {0, 2, 5, 8, 11, 14};

  for (int i = 0; i < 6; ++i) {
    const float* cx = ctx + (size_t)((i * NB + n) * 20) * HW + hw;
    float c0[10], fd[10];
    for (int q = 0; q < 10; ++q) { c0[q] = cx[q * HW]; fd[q] = cx[(10 + q) * HW]; }
    int cn = CNT[i];
    float a[3];
    for (int c = 0; c < cn; ++c) {
      float s = att_b[i * 3 + c];
      for (int q = 0; q < 10; ++q) s += att_w[(i * 3 + c) * 10 + q] * fd[q];
      a[c] = s;
      out_att[(n * 16 + COFF[i] + c) * HW + hw] = s;
    }
    float mx = a[0];
    for (int c = 1; c < cn; ++c) mx = fmaxf(mx, a[c]);
    float den = 0.f;
    for (int c = 0; c < cn; ++c) { a[c] = __expf(a[c] - mx); den += a[c]; }
    for (int c = 0; c < cn; ++c)
      att_soft[((i * NB + n) * 3 + c) * HW + hw] = a[c] / den;

    float o0 = ctx_b[i * 2 + 0], o1 = ctx_b[i * 2 + 1];
    for (int q = 0; q < 10; ++q) {
      o0 += ctx_w[(i * 2 + 0) * 10 + q] * c0[q];
      o1 += ctx_w[(i * 2 + 1) * 10 + q] * fd[q];
    }
    out_ctx[(n * 12 + 2 * i) * HW + hw]     = o0;
    out_ctx[(n * 12 + 2 * i + 1) * HW + hw] = o1;
  }

  float h0[10], h1[10];
  for (int q = 0; q < 10; ++q) {
    h0[q] = xh0[(n * 10 + q) * HW + hw];
    h1[q] = xh1[(n * 10 + q) * HW + hw];
  }
  {   // pu: 5 channels
    float pm[5];
    for (int c = 0; c < 5; ++c) {
      float s = du_b[c];
      for (int q = 0; q < 10; ++q) s += du_w[c * 10 + q] * h0[q];
      pm[c] = s;
      out_pu[(n * 5 + c) * HW + hw] = s;
    }
    float mx = pm[0];
    for (int c = 1; c < 5; ++c) mx = fmaxf(mx, pm[c]);
    float den = 0.f;
    for (int c = 0; c < 5; ++c) { pm[c] = __expf(pm[c] - mx); den += pm[c]; }
    for (int c = 0; c < 5; ++c) pu_att[(n * 5 + c) * HW + hw] = pm[c] / den;
  }
  {   // pl: 3 channels
    float pm[3];
    for (int c = 0; c < 3; ++c) {
      float s = dl_b[c];
      for (int q = 0; q < 10; ++q) s += dl_w[c * 10 + q] * h1[q];
      pm[c] = s;
      out_pl[(n * 3 + c) * HW + hw] = s;
    }
    float mx = pm[0];
    for (int c = 1; c < 3; ++c) mx = fmaxf(mx, pm[c]);
    float den = 0.f;
    for (int c = 0; c < 3; ++c) { pm[c] = __expf(pm[c] - mx); den += pm[c]; }
    for (int c = 0; c < 3; ++c) pl_att[(n * 3 + c) * HW + hw] = pm[c] / den;
  }
}

// ------------------------------------------------------------- graph+GRU
__device__ __forceinline__ void block2_dev(
    const float in[20],
    const float* __restrict__ c1w, const float* __restrict__ c1g, const float* __restrict__ c1b,
    const float* __restrict__ c2w, const float* __restrict__ c2g, const float* __restrict__ c2b,
    float out[10])
{
  float t[20];
  for (int o = 0; o < 20; ++o) {
    float s = 0.f;
    for (int c = 0; c < 20; ++c) s += c1w[o * 20 + c] * in[c];
    s = s * (c1g[o] * BNINV) + c1b[o];
    t[o] = s > 0.f ? s : 0.f;
  }
  for (int o = 0; o < 10; ++o) {
    float s = 0.f;
    for (int c = 0; c < 20; ++c) s += c2w[o * 20 + c] * t[c];
    s = s * (c2g[o] * BNINV) + c2b[o];
    out[o] = s > 0.f ? s : 0.f;
  }
}

__global__ __launch_bounds__(256) void k_graph(
    const float* __restrict__ ctx,
    const float* __restrict__ att_soft,
    const float* __restrict__ pu_att, const float* __restrict__ pl_att,
    const float* __restrict__ xh0, const float* __restrict__ xh1,
    const float* __restrict__ xp0, const float* __restrict__ xp1,
    const float* __restrict__ xp2, const float* __restrict__ xp3,
    const float* __restrict__ xp4, const float* __restrict__ xp5,
    const float* __restrict__ c1w, const float* __restrict__ c1g, const float* __restrict__ c1b,
    const float* __restrict__ c2w, const float* __restrict__ c2g, const float* __restrict__ c2b,
    const float* __restrict__ ggw, const float* __restrict__ ggb,
    const float* __restrict__ gcw, const float* __restrict__ gcb,
    float* __restrict__ out0)
{
  int p = blockIdx.x * 256 + threadIdx.x;
  if (p >= PTOT) return;
  int n = p / HW, hw = p % HW;

  const float* xpl[6] = {xp0, xp1, xp2, xp3, xp4, xp5};
  float h[6][10], msg[6][10];
  for (int i = 0; i < 6; ++i)
    for (int q = 0; q < 10; ++q) {
      h[i][q] = xpl[i][(n * 10 + q) * HW + hw];
      msg[i][q] = 0.f;
    }

  float xh0v[10], xh1v[10];
  for (int q = 0; q < 10; ++q) {
    xh0v[q] = xh0[(n * 10 + q) * HW + hw];
    xh1v[q] = xh1[(n * 10 + q) * HW + hw];
  }

  // dec_pu (blk set 0)
  for (int j = 0; j < 4; ++j) {
    float in[20], o[10];
    float att = pu_att[(n * 5 + j + 1) * HW + hw];
    for (int q = 0; q < 10; ++q) { in[q] = xh0v[q] * att; in[10 + q] = h[j][q]; }
    block2_dev(in, c1w, c1g, c1b, c2w, c2g, c2b, o);
    for (int q = 0; q < 10; ++q) msg[j][q] += o[q];
  }
  // dec_pl (blk set 1)
  for (int j = 0; j < 2; ++j) {
    float in[20], o[10];
    float att = pl_att[(n * 3 + j + 1) * HW + hw];
    for (int q = 0; q < 10; ++q) { in[q] = xh1v[q] * att; in[10 + q] = h[4 + j][q]; }
    block2_dev(in, c1w + 400, c1g + 20, c1b + 20, c2w + 200, c2g + 10, c2b + 10, o);
    for (int q = 0; q < 10; ++q) msg[4 + j][q] += o[q];
  }
  // edges (blk set 2)
  const int EU[10] = {0, 1, 1, 2, 2, 3, 3, 4, 4, 5};
  const int EV[10] = {1, 0, 2, 1, 3, 2, 4, 3, 5, 4};
  const int EK[10] = {1, 1, 2, 1, 2, 1, 2, 1, 2, 1};
  for (int e = 0; e < 10; ++e) {
    int u = EU[e], v = EV[e], k = EK[e];
    float att = att_soft[((u * NB + n) * 3 + k) * HW + hw];
    float in[20], o[10];
    const float* fd = ctx + (size_t)((u * NB + n) * 20 + 10) * HW + hw;
    for (int q = 0; q < 10; ++q) { in[q] = att * fd[q * HW]; in[10 + q] = h[v][q]; }
    block2_dev(in, c1w + 800, c1g + 40, c1b + 40, c2w + 400, c2g + 20, c2b + 20, o);
    for (int q = 0; q < 10; ++q) msg[v][q] += o[q];
  }
  // GRU per node
  for (int i = 0; i < 6; ++i) {
    float g[20];
    for (int o = 0; o < 20; ++o) {
      float s = ggb[i * 20 + o];
      const float* w = ggw + (i * 20 + o) * 20;
      for (int c = 0; c < 10; ++c) s += w[c] * msg[i][c] + w[10 + c] * h[i][c];
      g[o] = 1.f / (1.f + __expf(-s));
    }
    float cd[10];
    for (int o = 0; o < 10; ++o) {
      float s = gcb[i * 10 + o];
      const float* w = gcw + (i * 10 + o) * 20;
      for (int c = 0; c < 10; ++c) s += w[c] * msg[i][c] + w[10 + c] * (g[c] * h[i][c]);
      cd[o] = tanhf(s);
    }
    for (int c = 0; c < 10; ++c) {
      float z = g[10 + c];
      out0[(n * 60 + i * 10 + c) * HW + hw] = (1.f - z) * h[i][c] + z * cd[c];
    }
  }
}

// ---------------------------------------------------------------- launch
extern "C" void kernel_launch(void* const* d_in, const int* in_sizes, int n_in,
                              void* d_out, int out_size, void* d_ws, size_t ws_size,
                              hipStream_t stream) {
  const float* xh0 = (const float*)d_in[1];
  const float* xh1 = (const float*)d_in[2];
  const float* xp0 = (const float*)d_in[3];
  const float* xp1 = (const float*)d_in[4];
  const float* xp2 = (const float*)d_in[5];
  const float* xp3 = (const float*)d_in[6];
  const float* xp4 = (const float*)d_in[7];
  const float* xp5 = (const float*)d_in[8];
  const float* xp  = (const float*)d_in[9];
  const float* p1w = (const float*)d_in[10];
  const float* p1g = (const float*)d_in[11];
  const float* p1b = (const float*)d_in[12];
  const float* p2w = (const float*)d_in[13];
  const float* p2g = (const float*)d_in[14];
  const float* p2b = (const float*)d_in[15];
  const float* aw  = (const float*)d_in[16];
  const float* ab  = (const float*)d_in[17];
  const float* cw  = (const float*)d_in[18];
  const float* cb  = (const float*)d_in[19];
  const float* duw = (const float*)d_in[20];
  const float* dub = (const float*)d_in[21];
  const float* dlw = (const float*)d_in[22];
  const float* dlb = (const float*)d_in[23];
  const float* c1w = (const float*)d_in[24];
  const float* c1g = (const float*)d_in[25];
  const float* c1b = (const float*)d_in[26];
  const float* c2w = (const float*)d_in[27];
  const float* c2g = (const float*)d_in[28];
  const float* c2b = (const float*)d_in[29];
  const float* ggw = (const float*)d_in[30];
  const float* ggb = (const float*)d_in[31];
  const float* gcw = (const float*)d_in[32];
  const float* gcb = (const float*)d_in[33];
  float* out = (float*)d_out;

  // workspace layout (bytes)
  char* ws = (char*)d_ws;
  const size_t XP_E = (size_t)NB * CIN * HW;   // 14,745,600
  const size_t W1_E = 6u * 256u * 256u;        // 393,216
  const size_t W2_E = 6u * 20u * 256u;         // 30,720
  unsigned short* xpbf = (unsigned short*)ws;                    ws += XP_E * 2;
  unsigned short* w1bf = (unsigned short*)ws;                    ws += W1_E * 2;
  unsigned short* w2bf = (unsigned short*)ws;                    ws += W2_E * 2;
  float* ctx  = (float*)ws;  ws += (size_t)6 * NB * 20 * HW * 4;   // 27.6 MB
  float* asft = (float*)ws;  ws += (size_t)6 * NB * 3 * HW * 4;
  float* puat = (float*)ws;  ws += (size_t)NB * 5 * HW * 4;
  float* plat = (float*)ws;  ws += (size_t)NB * 3 * HW * 4;

  // 1) fp32 -> bf16 conversions
  k_f32_to_bf16<<<(int)((XP_E + 255) / 256), 256, 0, stream>>>(xp, xpbf, (int)XP_E);
  k_f32_to_bf16<<<(int)((W1_E + 255) / 256), 256, 0, stream>>>(p1w, w1bf, (int)W1_E);
  k_f32_to_bf16<<<(int)((W2_E + 255) / 256), 256, 0, stream>>>(p2w, w2bf, (int)W2_E);

  // 2) fused proj1+proj2 WMMA GEMM
  k_proj_fused<<<6 * NB * 75, 256, 0, stream>>>(xpbf, w1bf, w2bf, p1g, p1b, p2g, p2b, ctx);

  // 3) attention heads / softmaxes
  k_heads<<<(PTOT + 255) / 256, 256, 0, stream>>>(ctx, xh0, xh1, aw, ab, cw, cb,
                                                  duw, dub, dlw, dlb,
                                                  asft, puat, plat, out);

  // 4) decode blocks, edge messages, GRU
  k_graph<<<(PTOT + 255) / 256, 256, 0, stream>>>(ctx, asft, puat, plat, xh0, xh1,
                                                  xp0, xp1, xp2, xp3, xp4, xp5,
                                                  c1w, c1g, c1b, c2w, c2g, c2b,
                                                  ggw, ggb, gcw, gcb, out);
}
